// CrossFrameAttention_18897856103184
// MI455X (gfx1250) — compile-verified, matrix operand order checked by reading, not run on hardware
//
#include <hip/hip_runtime.h>
#include <hip/hip_bf16.h>

// ---------------------------------------------------------------------------
// CrossFrameAttention for MI455X (gfx1250): windowed cross attention + RoPE +
// softmax + output projection + LayerNorm + residual, one workgroup per 8x8
// window, all matmuls via v_wmma_f32_16x16x32_f16 (f16 in, f32 accumulate).
// ---------------------------------------------------------------------------

typedef _Float16 v16h  __attribute__((ext_vector_type(16)));
typedef _Float16 half8 __attribute__((ext_vector_type(8)));
typedef float    v8f   __attribute__((ext_vector_type(8)));

#define CC    192
#define HH    256
#define WWID  256
#define HEADS 4
#define HD    48
#define HDP   64          // head dim padded to 64 (zero filled) for K-loop of 2x32
#define NTOK  64          // tokens per window (8x8)
#define LDX   200         // row stride (f16) for X / O buffers (192 + pad, mult of 8)
#define LDQ   264         // row stride (f16) for Q/K/V buffers (4*64 + pad, mult of 8)
#define LDP   72          // row stride (f16) for P buffer (64 + pad, mult of 8)
#define LDY   196         // row stride (f32) for Y buffer
#define SCALE_F 0.14433756729740643f   // 48^-0.5
#define EPS_F   1e-5f

// LDS layout (bytes). Regions alias across phases:
//   [0      .. 25600)  sXq  (f16 64x200)   | later: sS (f32 64x64 @0) + sP (f16 64x72 @16384)
//   [25600  .. 51200)  sXkv (f16 64x200)   | later: sO (f16 64x200)
//   [51200  .. 152576) sQKV (f16 3x64x264) | later: sY (f32 64x196 @51200) + mu/rs (@101376/101632)
#define OFF_XKV   25600
#define OFF_QKV   51200
#define OFF_P     16384
#define OFF_MU    101376
#define OFF_RS    101632
#define QKV_ONE   16896      // 64*264 f16 elements per tensor
#define SMEM_BYTES 152576

union FragU { v16h f; half8 h[2]; };

// Load a 16-element f16 fragment whose K index is contiguous in memory.
// Per-lane element e maps to k = 16*(e>>3) + 8*half + (e&7)  (matches the ISA
// 16-bit A 16x32 table; B 32x16 uses the same per-lane pattern with col=lane&15).
__device__ __forceinline__ v16h load_frag_k(const _Float16* base, int half) {
  FragU u;
  u.h[0] = *reinterpret_cast<const half8*>(base + 8 * half);
  u.h[1] = *reinterpret_cast<const half8*>(base + 16 + 8 * half);
  return u.f;
}

__device__ __forceinline__ v8f wmma32(v16h a, v16h b, v8f c) {
  return __builtin_amdgcn_wmma_f32_16x16x32_f16(false, a, false, b, (short)0, c,
                                                false, false);
}

// ---------------------------------------------------------------------------
// Prep kernel: convert Wq/Wk/Wv/Wo (f32, 192x192 each) to f16 in workspace and
// build the RoPE cos/sin table: angle(n,d) = (i+j) * 10000^(-2d/48), n=i*8+j.
// ---------------------------------------------------------------------------
__global__ void cfa_prep_kernel(const float* __restrict__ Wq,
                                const float* __restrict__ Wk,
                                const float* __restrict__ Wv,
                                const float* __restrict__ Wo,
                                _Float16* __restrict__ wgt,
                                float* __restrict__ rcos,
                                float* __restrict__ rsin) {
  int g = blockIdx.x * blockDim.x + threadIdx.x;
  const int NW = 4 * CC * CC;       // 147456
  if (g < NW) {
    int w = g / (CC * CC);
    int i = g % (CC * CC);
    const float* src = (w == 0) ? Wq : (w == 1) ? Wk : (w == 2) ? Wv : Wo;
    wgt[g] = (_Float16)src[i];
  } else {
    int e = g - NW;
    if (e < NTOK * (HD / 2)) {      // 64 * 24
      int n = e / 24, d = e % 24;
      float inv = powf(10000.0f, -2.0f * (float)d / 48.0f);
      float ang = (float)((n >> 3) + (n & 7)) * inv;
      rcos[e] = cosf(ang);
      rsin[e] = sinf(ang);
    }
  }
}

// ---------------------------------------------------------------------------
// Main kernel: one block (8 wave32) per window.
// ---------------------------------------------------------------------------
__global__ void __launch_bounds__(256)
cfa_attn_kernel(const float* __restrict__ temporal,
                const float* __restrict__ spatial,
                const _Float16* __restrict__ wgt,
                const float* __restrict__ rcos,
                const float* __restrict__ rsin,
                const float* __restrict__ gamma,
                const float* __restrict__ beta,
                float* __restrict__ out) {
  extern __shared__ char smem[];
  _Float16* sXq  = (_Float16*)(smem);
  _Float16* sXkv = (_Float16*)(smem + OFF_XKV);
  _Float16* sQKV = (_Float16*)(smem + OFF_QKV);
  float*    sS   = (float*)(smem);                 // alias of sXq (post-proj)
  _Float16* sP   = (_Float16*)(smem + OFF_P);      // alias of sXq tail
  _Float16* sO   = (_Float16*)(smem + OFF_XKV);    // alias of sXkv (post-proj)
  float*    sY   = (float*)(smem + OFF_QKV);       // alias of sQKV (post-attn)
  float*    sMu  = (float*)(smem + OFF_MU);
  float*    sRs  = (float*)(smem + OFF_RS);

  const int t    = threadIdx.x;
  const int half = (t >> 4) & 1;    // lane/16 within wave32
  const int col  = t & 15;          // lane%16
  const int wv   = t >> 5;          // wave id 0..7

  const int blk = blockIdx.x;       // 4096 windows
  const int b   = blk >> 10;
  const int rem = blk & 1023;
  const int h0  = (rem >> 5) * 8;
  const int w0  = (rem & 31) * 8;

  // --- Phase 0: zero the padded Q/K/V region, stage both windows as f16 -----
  for (int e = t; e < (3 * QKV_ONE * 2) / 16; e += 256)
    ((float4*)(smem + OFF_QKV))[e] = make_float4(0.f, 0.f, 0.f, 0.f);

  for (int e = t; e < CC * NTOK; e += 256) {
    int c = e >> 6, n = e & 63;
    long g = ((long)(b * CC + c) * HH + h0 + (n >> 3)) * WWID + w0 + (n & 7);
    sXq[n * LDX + c]  = (_Float16)temporal[g];
    sXkv[n * LDX + c] = (_Float16)spatial[g];
  }
  __syncthreads();

  // --- Phase 1: Q/K/V projections: 3 * (64x192 = [64x192]@[192x192]^T) ------
  // 144 tiles of 16x16, K = 6 steps of 32. D lane owns (M = r+8*half, N = col).
  for (int tile = wv; tile < 144; tile += 8) {
    int proj = tile / 48, r2 = tile % 48, tm = r2 / 12, tn = r2 % 12;
    const _Float16* A  = proj ? sXkv : sXq;
    const _Float16* Wt = wgt + proj * (CC * CC);
    v8f acc = {};
#pragma unroll
    for (int kt = 0; kt < 6; ++kt) {
      v16h a  = load_frag_k(A + (tm * 16 + col) * LDX + kt * 32, half);
      v16h bm = load_frag_k(Wt + (tn * 16 + col) * CC + kt * 32, half);
      acc = wmma32(a, bm, acc);
    }
    int co = tn * 16 + col, hh = co / HD, dd = co % HD;
    _Float16* dst = sQKV + proj * QKV_ONE + hh * HDP + dd;
#pragma unroll
    for (int r = 0; r < 8; ++r)
      dst[(tm * 16 + half * 8 + r) * LDQ] = (_Float16)acc[r];
  }
  __syncthreads();

  // --- Phase 2: RoPE on Q and K (scale folded into Q) -----------------------
  for (int e = t; e < 2 * NTOK * HEADS * 24; e += 256) {
    int which = e / 6144, r2 = e % 6144;
    int n = r2 / 96, r3 = r2 % 96, hh = r3 / 24, dd = r3 % 24;
    _Float16* base = sQKV + which * QKV_ONE + n * LDQ + hh * HDP;
    float c0 = rcos[n * 24 + dd], s0 = rsin[n * 24 + dd];
    float x1 = (float)base[dd], x2 = (float)base[dd + 24];
    float sc = which ? 1.0f : SCALE_F;
    base[dd]      = (_Float16)((x1 * c0 - x2 * s0) * sc);
    base[dd + 24] = (_Float16)((x1 * s0 + x2 * c0) * sc);
  }
  __syncthreads();

  const _Float16* sQ = sQKV;
  const _Float16* sK = sQKV + QKV_ONE;
  const _Float16* sV = sQKV + 2 * QKV_ONE;

  // --- Phase 3: attention, one head at a time across all 8 waves ------------
  for (int h = 0; h < HEADS; ++h) {
    // S = Q_h @ K_h^T  (64x64, K = 64 padded head dim = 2x32)
    for (int tile = wv; tile < 16; tile += 8) {
      int tm = tile >> 2, tn = tile & 3;
      v8f acc = {};
#pragma unroll
      for (int kt = 0; kt < 2; ++kt) {
        v16h a  = load_frag_k(sQ + (tm * 16 + col) * LDQ + h * HDP + kt * 32, half);
        v16h bm = load_frag_k(sK + (tn * 16 + col) * LDQ + h * HDP + kt * 32, half);
        acc = wmma32(a, bm, acc);
      }
#pragma unroll
      for (int r = 0; r < 8; ++r)
        sS[(tm * 16 + half * 8 + r) * 64 + tn * 16 + col] = acc[r];
    }
    __syncthreads();

    // Row softmax (64 rows), write P as f16.
    if (t < 64) {
      const float* row = sS + t * 64;
      float mx = -3.4e38f;
      for (int m = 0; m < 64; ++m) mx = fmaxf(mx, row[m]);
      float sum = 0.f;
      for (int m = 0; m < 64; ++m) sum += __expf(row[m] - mx);
      float inv = 1.0f / sum;
      _Float16* prow = sP + t * LDP;
      for (int m = 0; m < 64; ++m)
        prow[m] = (_Float16)(__expf(row[m] - mx) * inv);
    }
    __syncthreads();

    // O_h = P @ V_h  (64x48, K over 64 tokens = 2x32). B is a token-major
    // gather from LDS (k = token index -> per-element loads).
    for (int tile = wv; tile < 12; tile += 8) {
      int tm = tile / 3, tn = tile % 3;
      v8f acc = {};
#pragma unroll
      for (int kt = 0; kt < 2; ++kt) {
        v16h a = load_frag_k(sP + (tm * 16 + col) * LDP + kt * 32, half);
        v16h bv;
#pragma unroll
        for (int e = 0; e < 16; ++e) {
          int k = ((e >> 3) << 4) | (half << 3) | (e & 7);
          bv[e] = sV[(kt * 32 + k) * LDQ + h * HDP + tn * 16 + col];
        }
        acc = wmma32(a, bv, acc);
      }
#pragma unroll
      for (int r = 0; r < 8; ++r)
        sO[(tm * 16 + half * 8 + r) * LDX + h * HD + tn * 16 + col] =
            (_Float16)acc[r];
    }
    __syncthreads();
  }

  // --- Phase 4: output projection Y = O @ Wo^T (f32 result in LDS) ----------
  const _Float16* Wo = wgt + 3 * (CC * CC);
  for (int tile = wv; tile < 48; tile += 8) {
    int tm = tile / 12, tn = tile % 12;
    v8f acc = {};
#pragma unroll
    for (int kt = 0; kt < 6; ++kt) {
      v16h a  = load_frag_k(sO + (tm * 16 + col) * LDX + kt * 32, half);
      v16h bm = load_frag_k(Wo + (tn * 16 + col) * CC + kt * 32, half);
      acc = wmma32(a, bm, acc);
    }
#pragma unroll
    for (int r = 0; r < 8; ++r)
      sY[(tm * 16 + half * 8 + r) * LDY + tn * 16 + col] = acc[r];
  }
  __syncthreads();

  // --- Phase 5: LayerNorm stats per token -----------------------------------
  if (t < 64) {
    const float* row = sY + t * LDY;
    float s1 = 0.f, s2 = 0.f;
    for (int c = 0; c < CC; ++c) { float v = row[c]; s1 += v; s2 += v * v; }
    float mu  = s1 * (1.0f / CC);
    float var = s2 * (1.0f / CC) - mu * mu;
    sMu[t] = mu;
    sRs[t] = rsqrtf(var + EPS_F);
  }
  __syncthreads();

  // --- Phase 6: normalize + affine + residual, coalesced scatter back -------
  for (int e = t; e < CC * NTOK; e += 256) {
    int c = e >> 6, n = e & 63;
    long g = ((long)(b * CC + c) * HH + h0 + (n >> 3)) * WWID + w0 + (n & 7);
    float y = sY[n * LDY + c];
    out[g] = (y - sMu[n]) * sRs[n] * gamma[c] + beta[c] + temporal[g];
  }
}

// ---------------------------------------------------------------------------
extern "C" void kernel_launch(void* const* d_in, const int* in_sizes, int n_in,
                              void* d_out, int out_size, void* d_ws,
                              size_t ws_size, hipStream_t stream) {
  const float* temporal = (const float*)d_in[0];
  const float* spatial  = (const float*)d_in[1];
  const float* Wq       = (const float*)d_in[2];
  const float* Wk       = (const float*)d_in[3];
  const float* Wv       = (const float*)d_in[4];
  const float* Wo       = (const float*)d_in[5];
  const float* gamma    = (const float*)d_in[6];
  const float* beta     = (const float*)d_in[7];

  // Workspace: 4 f16 weight matrices (294912 B) + rope cos/sin (2*6144 B).
  _Float16* wgt  = (_Float16*)d_ws;
  float*    rcos = (float*)((char*)d_ws + 4 * CC * CC * 2);
  float*    rsin = rcos + NTOK * 24;

  int prepN = 4 * CC * CC + NTOK * 24;  // 148992
  cfa_prep_kernel<<<(prepN + 255) / 256, 256, 0, stream>>>(Wq, Wk, Wv, Wo, wgt,
                                                           rcos, rsin);

  (void)hipFuncSetAttribute((const void*)cfa_attn_kernel,
                            hipFuncAttributeMaxDynamicSharedMemorySize,
                            SMEM_BYTES);
  cfa_attn_kernel<<<4096, 256, SMEM_BYTES, stream>>>(
      temporal, spatial, wgt, rcos, rsin, gamma, beta, (float*)d_out);
}
